// PointNetSetAbstraction_6416681141059
// MI455X (gfx1250) — matrix operand order repaired and checked by reading, not compiled, not run on hardware
//
#include <hip/hip_runtime.h>
#include <hip/hip_bf16.h>

// ---------------------------------------------------------------------------
// PointNet++ Set Abstraction for MI455X (gfx1250, wave32, WMMA)
// Pipeline: FPS -> ball query -> gather(concat) -> 3x [GEMM(WMMA f16, async
// LDS double-buffered) + BN stats + norm/ReLU] -> 64-way maxpool.
// Bias is dropped (cancels under global batchnorm).
// ---------------------------------------------------------------------------

typedef _Float16 h16;
typedef __attribute__((ext_vector_type(16))) _Float16 v16h;
typedef __attribute__((ext_vector_type(8)))  float    v8f;

#define BATCH   16
#define NPTS    4096
#define S_PTS   512
#define K_NBR   64
#define D_FEAT  128
#define C0      131        // D_FEAT + 3
#define C0P     160        // padded to multiple of 32
#define MROWS   (BATCH * S_PTS * K_NBR)   // 524288
#define R2      0.16f
#define XYZ_OUT (BATCH * 3 * S_PTS)       // 24576 floats

// ---------------------------------------------------------------------------
// Weight conversion fp32 -> fp16 with K padding
// ---------------------------------------------------------------------------
__global__ void wconv_kernel(const float* __restrict__ w, h16* __restrict__ o,
                             int OC, int IC, int KP) {
  int i = blockIdx.x * blockDim.x + threadIdx.x;
  if (i >= OC * KP) return;
  int oc = i / KP, k = i - oc * KP;
  o[i] = (k < IC) ? (h16)w[oc * IC + k] : (h16)0.f;
}

// ---------------------------------------------------------------------------
// Farthest point sampling: one block per batch, dist array in LDS
// ---------------------------------------------------------------------------
__global__ __launch_bounds__(512) void fps_kernel(const float* __restrict__ xyz,
                                                  int* __restrict__ fps_idx,
                                                  float* __restrict__ newxyz,
                                                  float* __restrict__ out) {
  int b = blockIdx.x;
  int t = threadIdx.x;
  const float* X = xyz + (size_t)b * 3 * NPTS;
  __shared__ float dist[NPTS];
  __shared__ float sval[512];
  __shared__ int   sidx[512];
  for (int j = t; j < NPTS; j += 512) dist[j] = 1e10f;
  int far = 0;
  __syncthreads();
  for (int i = 0; i < S_PTS; ++i) {
    float fx = X[far], fy = X[NPTS + far], fz = X[2 * NPTS + far];
    if (t == 0) fps_idx[b * S_PTS + i] = far;
    if (t < 3) {
      float v = X[t * NPTS + far];
      newxyz[((size_t)b * S_PTS + i) * 3 + t] = v;
      out[((size_t)b * 3 + t) * S_PTS + i] = v;   // new_xyz transposed (B,3,S)
    }
    float bv = -1.f; int bi = 0;
    for (int j = 0; j < 8; ++j) {
      int n = t * 8 + j;
      float dx = X[n] - fx, dy = X[NPTS + n] - fy, dz = X[2 * NPTS + n] - fz;
      float d = dx * dx + dy * dy + dz * dz;
      float dn = dist[n];
      d = d < dn ? d : dn;
      dist[n] = d;
      if (d > bv) { bv = d; bi = n; }
    }
    sval[t] = bv; sidx[t] = bi;
    __syncthreads();
    for (int s = 256; s > 0; s >>= 1) {
      if (t < s) {
        float v2 = sval[t + s]; int i2 = sidx[t + s];
        if (v2 > sval[t] || (v2 == sval[t] && i2 < sidx[t])) { sval[t] = v2; sidx[t] = i2; }
      }
      __syncthreads();
    }
    far = sidx[0];
    __syncthreads();
  }
}

// ---------------------------------------------------------------------------
// Ball query: one wave32 per centroid, in-order scan + ballot compaction
// ---------------------------------------------------------------------------
__global__ __launch_bounds__(256) void ball_query_kernel(const float* __restrict__ xyz,
                                                         const float* __restrict__ newxyz,
                                                         int* __restrict__ gidx) {
  int w = (blockIdx.x * 256 + threadIdx.x) >> 5;   // centroid id
  int lane = threadIdx.x & 31;
  int b = w >> 9;
  const float* X = xyz + (size_t)b * 3 * NPTS;
  const float* c = &newxyz[(size_t)w * 3];
  float cx = c[0], cy = c[1], cz = c[2];
  int* out = gidx + (size_t)w * K_NBR;
  int cnt = 0, first = -1;
  for (int n0 = 0; n0 < NPTS; n0 += 32) {
    int n = n0 + lane;
    float dx = X[n] - cx, dy = X[NPTS + n] - cy, dz = X[2 * NPTS + n] - cz;
    bool pred = (dx * dx + dy * dy + dz * dz) <= R2;
    unsigned mask = (unsigned)__ballot(pred);
    if (first < 0 && mask) first = n0 + __ffs(mask) - 1;
    if (pred) {
      int pos = cnt + __popc(mask & ((1u << lane) - 1u));
      if (pos < K_NBR) out[pos] = n;
    }
    cnt += __popc(mask);
    if (cnt >= K_NBR) break;
  }
  if (cnt < K_NBR) {
    int f = first < 0 ? 0 : first;
    for (int j = cnt + lane; j < K_NBR; j += 32) out[j] = f;
  }
}

// ---------------------------------------------------------------------------
// Gather grouped features + centered xyz into fp16 activation [M][C0P]
// ---------------------------------------------------------------------------
__global__ __launch_bounds__(256) void gather_kernel(const float* __restrict__ xyz,
                                                     const float* __restrict__ points,
                                                     const int* __restrict__ gidx,
                                                     const float* __restrict__ newxyz,
                                                     h16* __restrict__ A) {
  int g = blockIdx.x;                 // b*S + s
  int b = g >> 9;
  __shared__ int   sidx[K_NBR];
  __shared__ float cc[3];
  if (threadIdx.x < K_NBR) sidx[threadIdx.x] = gidx[(size_t)g * K_NBR + threadIdx.x];
  if (threadIdx.x < 3) cc[threadIdx.x] = newxyz[(size_t)g * 3 + threadIdx.x];
  __syncthreads();
  const float* ptsb = points + (size_t)b * D_FEAT * NPTS;
  const float* xyzb = xyz + (size_t)b * 3 * NPTS;
  for (int e = threadIdx.x; e < K_NBR * C0P; e += 256) {
    int k = e / C0P;
    int c = e - k * C0P;
    int n = sidx[k];
    float v;
    if (c < D_FEAT)       v = ptsb[(size_t)c * NPTS + n];
    else if (c < C0)      v = xyzb[(size_t)(c - D_FEAT) * NPTS + n] - cc[c - D_FEAT];
    else                  v = 0.f;
    A[((size_t)g * K_NBR + k) * C0P + c] = (h16)v;
  }
}

// ---------------------------------------------------------------------------
// WMMA GEMM: Y[m][n] = sum_k A[m][k] * W[n][k]  (fp16 in, fp32 acc, fp16 out)
// Block tile 128x128, 8 waves (4 in M x 2 in N), each wave 32x64 = 2x4 WMMA.
// Tiles are filled with GLOBAL_LOAD_ASYNC_TO_LDS_B128 into double-buffered
// LDS (ASYNCcnt tracked, no VGPR staging); per-channel sum / sum-of-squares
// for global batchnorm accumulated via LDS float atomics + one global atomic
// per channel per block.
// ---------------------------------------------------------------------------
#define A_STRIDE 40   // halfs per LDS row (padded to spread banks)
union Frag { uint4 q[2]; v16h v; };

// one 16-byte async copy global -> LDS (per lane)
__device__ __forceinline__ void async_copy_b128(unsigned lds_off, const void* gaddr) {
  asm volatile("global_load_async_to_lds_b128 %0, %1, off"
               :: "v"(lds_off), "v"(gaddr) : "memory");
}

// issue one 128x32-half tile of A and one of W (4 async instrs per wave)
__device__ __forceinline__ void issue_tile_async(const h16* __restrict__ A,
                                                 const h16* __restrict__ W,
                                                 h16* sA, h16* sB,
                                                 size_t m0, int n0, int Kpad,
                                                 int kk, int tid) {
#pragma unroll
  for (int i0 = 0; i0 < 512; i0 += 256) {
    int i = i0 + tid;
    int r = i >> 2, c4 = i & 3;
    async_copy_b128((unsigned)(uintptr_t)&sA[r * A_STRIDE + c4 * 8],
                    &A[(m0 + r) * (size_t)Kpad + kk + c4 * 8]);
    async_copy_b128((unsigned)(uintptr_t)&sB[r * A_STRIDE + c4 * 8],
                    &W[(size_t)(n0 + r) * Kpad + kk + c4 * 8]);
  }
}

__global__ __launch_bounds__(256) void gemm_bn_kernel(const h16* __restrict__ A,
                                                      const h16* __restrict__ W,
                                                      h16* __restrict__ Y,
                                                      float* __restrict__ gsum,
                                                      float* __restrict__ gsq,
                                                      int N, int Kpad) {
  __shared__ h16 sA[2][128 * A_STRIDE];
  __shared__ h16 sB[2][128 * A_STRIDE];
  __shared__ float s_sum[128];
  __shared__ float s_sq[128];

  int tid  = threadIdx.x;
  int lane = tid & 31;
  int wave = tid >> 5;
  int wm   = wave & 3;   // 4 waves along M, 32 rows each
  int wn   = wave >> 2;  // 2 waves along N, 64 cols each
  size_t m0 = (size_t)blockIdx.x * 128;
  int    n0 = blockIdx.y * 128;

  if (tid < 128) { s_sum[tid] = 0.f; s_sq[tid] = 0.f; }

  v8f acc[2][4] = {};

  int nk = Kpad >> 5;
  issue_tile_async(A, W, sA[0], sB[0], m0, n0, Kpad, 0, tid);

  for (int t = 0; t < nk; ++t) {
    int cur = t & 1;
    if (t + 1 < nk) {
      // prefetch next tile into the other buffer (safe: barrier at end of
      // previous iteration guarantees all waves finished reading it)
      issue_tile_async(A, W, sA[cur ^ 1], sB[cur ^ 1], m0, n0, Kpad,
                       (t + 1) << 5, tid);
      // async loads complete in order: current tile done when only the
      // 4 next-tile copies remain outstanding
      asm volatile("s_wait_asynccnt 0x4" ::: "memory");
    } else {
      asm volatile("s_wait_asynccnt 0x0" ::: "memory");
    }
    __syncthreads();   // tile visible to all waves

    // A fragments: lane<16 -> M=lane, K = 0..7 & 16..23 ; lane>=16 -> K+8
    Frag a[2];
    int ao = (lane >> 4) * 8;  // halfs
#pragma unroll
    for (int mt = 0; mt < 2; ++mt) {
      const char* ab = (const char*)&sA[cur][(wm * 32 + mt * 16 + (lane & 15)) * A_STRIDE];
      a[mt].q[0] = *(const uint4*)(ab + ao * 2);
      a[mt].q[1] = *(const uint4*)(ab + ao * 2 + 32);
    }
    // B fragments: lane<16 -> N=lane, K = 0..7 & 16..23 ; lane>=16 -> K+8
    Frag bf[4];
    int bo = (lane >> 4) * 8;  // halfs
#pragma unroll
    for (int nt = 0; nt < 4; ++nt) {
      const char* bb = (const char*)&sB[cur][(wn * 64 + nt * 16 + (lane & 15)) * A_STRIDE];
      bf[nt].q[0] = *(const uint4*)(bb + bo * 2);
      bf[nt].q[1] = *(const uint4*)(bb + bo * 2 + 32);
    }
#pragma unroll
    for (int mt = 0; mt < 2; ++mt)
#pragma unroll
      for (int nt = 0; nt < 4; ++nt)
        acc[mt][nt] = __builtin_amdgcn_wmma_f32_16x16x32_f16(
            false, a[mt].v, false, bf[nt].v, (short)0, acc[mt][nt], false, false);
    __syncthreads();   // all waves done with this buffer before refill
  }

  // store fp16 Y + accumulate batchnorm statistics
#pragma unroll
  for (int mt = 0; mt < 2; ++mt)
#pragma unroll
    for (int nt = 0; nt < 4; ++nt) {
      int nl = wn * 64 + nt * 16 + (lane & 15);
      int gn = n0 + nl;
      size_t gm = m0 + wm * 32 + mt * 16 + (lane >> 4) * 8;
      float s = 0.f, sq = 0.f;
#pragma unroll
      for (int r = 0; r < 8; ++r) {
        float v = acc[mt][nt][r];
        Y[(gm + r) * (size_t)N + gn] = (h16)v;
        s += v; sq += v * v;
      }
      atomicAdd(&s_sum[nl], s);     // ds_add_f32
      atomicAdd(&s_sq[nl], sq);
    }
  __syncthreads();
  if (tid < 128) {
    atomicAdd(&gsum[n0 + tid], s_sum[tid]);  // global_atomic_add_f32
    atomicAdd(&gsq[n0 + tid], s_sq[tid]);
  }
}

// ---------------------------------------------------------------------------
// BN helpers
// ---------------------------------------------------------------------------
__global__ void zero_stats_kernel(float* sum, float* sq, int n) {
  int i = blockIdx.x * blockDim.x + threadIdx.x;
  if (i < n) { sum[i] = 0.f; sq[i] = 0.f; }
}

__global__ void bn_finalize_kernel(const float* __restrict__ sum,
                                   const float* __restrict__ sq,
                                   const float* __restrict__ gamma,
                                   const float* __restrict__ beta,
                                   float* __restrict__ scale,
                                   float* __restrict__ shift,
                                   int n, float invM) {
  int c = blockIdx.x * blockDim.x + threadIdx.x;
  if (c >= n) return;
  float m  = sum[c] * invM;
  float v  = sq[c] * invM - m * m;
  float rs = rsqrtf(v + 1e-5f);
  float sc = gamma[c] * rs;
  scale[c] = sc;
  shift[c] = beta[c] - m * sc;
}

// y*scale + shift, ReLU, fp16 out; 8 halfs (B128) per thread.
// N must be a power of two (128 here).
union H8 { uint4 q; h16 h[8]; };
__global__ __launch_bounds__(256) void norm_relu_kernel(const h16* __restrict__ Y,
                                                        const float* __restrict__ scale,
                                                        const float* __restrict__ shift,
                                                        h16* __restrict__ A,
                                                        size_t total8, int Nmask) {
  size_t i = (size_t)blockIdx.x * 256 + threadIdx.x;
  if (i >= total8) return;
  size_t base = i * 8;
  int c0 = (int)(base & (size_t)Nmask);
  H8 u;
  u.q = *(const uint4*)(Y + base);
#pragma unroll
  for (int j = 0; j < 8; ++j) {
    float v = (float)u.h[j] * scale[c0 + j] + shift[c0 + j];
    u.h[j] = (h16)(v > 0.f ? v : 0.f);
  }
  *(uint4*)(A + base) = u.q;
}

// ---------------------------------------------------------------------------
// Fused BN + ReLU + max over K=64 neighbors, write final features (fp32)
// ---------------------------------------------------------------------------
__global__ __launch_bounds__(256) void maxpool_kernel(const h16* __restrict__ Y2,
                                                      const float* __restrict__ scale,
                                                      const float* __restrict__ shift,
                                                      float* __restrict__ out) {
  int g = blockIdx.x;            // b*S + s
  int o = threadIdx.x;           // 256 channels
  int b = g >> 9, s = g & 511;
  const h16* y = Y2 + (size_t)g * K_NBR * 256;
  float sc = scale[o], sh = shift[o];
  float m = 0.f;                 // ReLU output floor; max(relu) >= 0
  for (int k = 0; k < K_NBR; ++k) {
    float v = (float)y[(size_t)k * 256 + o] * sc + sh;
    v = v > 0.f ? v : 0.f;
    m = v > m ? v : m;
  }
  out[XYZ_OUT + ((size_t)b * 256 + o) * S_PTS + s] = m;
}

// ---------------------------------------------------------------------------
// Host launcher
// ---------------------------------------------------------------------------
extern "C" void kernel_launch(void* const* d_in, const int* in_sizes, int n_in,
                              void* d_out, int out_size, void* d_ws, size_t ws_size,
                              hipStream_t stream) {
  const float* xyz    = (const float*)d_in[0];
  const float* points = (const float*)d_in[1];
  const float* w0 = (const float*)d_in[2];
  const float* g0 = (const float*)d_in[4];
  const float* be0 = (const float*)d_in[5];
  const float* w1 = (const float*)d_in[6];
  const float* g1 = (const float*)d_in[8];
  const float* be1 = (const float*)d_in[9];
  const float* w2 = (const float*)d_in[10];
  const float* g2 = (const float*)d_in[12];
  const float* be2 = (const float*)d_in[13];
  float* out = (float*)d_out;

  char* base = (char*)d_ws;
  size_t off = 0;
  auto alloc = [&](size_t bytes) -> char* {
    char* p = base + off;
    off = (off + bytes + 255) & ~(size_t)255;
    return p;
  };
  int*   fps_idx = (int*)  alloc((size_t)BATCH * S_PTS * 4);
  int*   gidx    = (int*)  alloc((size_t)BATCH * S_PTS * K_NBR * 4);
  float* newxyz  = (float*)alloc((size_t)BATCH * S_PTS * 3 * 4);
  h16*   w0h     = (h16*)  alloc((size_t)128 * C0P * 2);
  h16*   w1h     = (h16*)  alloc((size_t)128 * 128 * 2);
  h16*   w2h     = (h16*)  alloc((size_t)256 * 128 * 2);
  float* gsum    = (float*)alloc(256 * 4);
  float* gsq     = (float*)alloc(256 * 4);
  float* scale   = (float*)alloc(256 * 4);
  float* shift   = (float*)alloc(256 * 4);
  h16*   bufA    = (h16*)  alloc((size_t)MROWS * C0P * 2);   // activations in
  h16*   bufB    = (h16*)  alloc((size_t)MROWS * 256 * 2);   // GEMM out

  // weights -> fp16 (padded)
  wconv_kernel<<<(128 * C0P + 255) / 256, 256, 0, stream>>>(w0, w0h, 128, C0, C0P);
  wconv_kernel<<<(128 * 128 + 255) / 256, 256, 0, stream>>>(w1, w1h, 128, 128, 128);
  wconv_kernel<<<(256 * 128 + 255) / 256, 256, 0, stream>>>(w2, w2h, 256, 128, 128);

  // sampling / grouping
  fps_kernel<<<BATCH, 512, 0, stream>>>(xyz, fps_idx, newxyz, out);
  ball_query_kernel<<<(BATCH * S_PTS * 32) / 256, 256, 0, stream>>>(xyz, newxyz, gidx);
  gather_kernel<<<BATCH * S_PTS, 256, 0, stream>>>(xyz, points, gidx, newxyz, bufA);

  const float invM = 1.0f / (float)MROWS;
  const int   mblk = MROWS / 128;   // 4096

  struct Layer { const h16* W; const float* g; const float* b; int N; int Kpad; };
  Layer layers[3] = { { w0h, g0, be0, 128, C0P },
                      { w1h, g1, be1, 128, 128 },
                      { w2h, g2, be2, 256, 128 } };

  for (int l = 0; l < 3; ++l) {
    int N = layers[l].N, Kpad = layers[l].Kpad;
    zero_stats_kernel<<<1, 256, 0, stream>>>(gsum, gsq, N);
    gemm_bn_kernel<<<dim3(mblk, N / 128), 256, 0, stream>>>(
        bufA, layers[l].W, bufB, gsum, gsq, N, Kpad);
    bn_finalize_kernel<<<1, 256, 0, stream>>>(gsum, gsq, layers[l].g, layers[l].b,
                                              scale, shift, N, invM);
    if (l < 2) {
      size_t total8 = (size_t)MROWS * N / 8;
      norm_relu_kernel<<<(unsigned)((total8 + 255) / 256), 256, 0, stream>>>(
          bufB, scale, shift, bufA, total8, N - 1);
    }
  }
  maxpool_kernel<<<BATCH * S_PTS, 256, 0, stream>>>(bufB, scale, shift, out);
}
